// MultiHeadAttention_80006650790252
// MI455X (gfx1250) — compile-verified, matrix operand order checked by reading, not run on hardware
//
#include <hip/hip_runtime.h>

#define B_ 8
#define NT_ 512
#define NC_ 2048
#define EMB_ 512
#define H_ 8
#define DK_ 64
#define DV_ 64

typedef __attribute__((ext_vector_type(16))) _Float16 v16h;
typedef __attribute__((ext_vector_type(8)))  float    v8f;

// ---- WMMA fragment loaders -------------------------------------------------
// CDNA5 V_WMMA_F32_16X16X32_F16 A-operand layout (ISA 7.12.2):
//   lane L: row = L%16, hi = L/16
//   halves a[0..7]  = src[row][hi*8 + 0..7]
//   halves a[8..15] = src[row][16 + hi*8 + 0..7]
// B-operand (K x N) is symmetric with lane holding column n = L%16, so loading
// rows of a row-major (N x K) matrix with the same pattern yields B = thatᵀ.

__device__ __forceinline__ v16h frag_from_f32(const float* __restrict__ src, int ld) {
  const int lane = threadIdx.x & 31;
  const int row  = lane & 15;
  const int hi   = lane >> 4;
  const float* p = src + (size_t)row * ld + hi * 8;
  v16h a;
#pragma unroll
  for (int j = 0; j < 8; ++j) a[j] = (_Float16)p[j];
#pragma unroll
  for (int j = 0; j < 8; ++j) a[8 + j] = (_Float16)p[16 + j];
  return a;
}

__device__ __forceinline__ v16h frag_from_f16(const _Float16* __restrict__ src, int ld) {
  const int lane = threadIdx.x & 31;
  const int row  = lane & 15;
  const int hi   = lane >> 4;
  const _Float16* p = src + (size_t)row * ld + hi * 8;
  v16h a;
#pragma unroll
  for (int j = 0; j < 8; ++j) a[j] = p[j];
#pragma unroll
  for (int j = 0; j < 8; ++j) a[8 + j] = p[16 + j];
  return a;
}

__device__ __forceinline__ v8f wmma_step(v16h a, v16h b, v8f c) {
  return __builtin_amdgcn_wmma_f32_16x16x32_f16(false, a, false, b, (short)0, c,
                                                false, false);
}

// ---- QKV projection: Y = X @ W^T + bias, reshaped to heads ----------------
// X: (M, 512) f32, W: (512, 512) f32 (row = out-feature), out f16.
// TRANS_OUT=false: out[(b*H+h)*rows + r][d]      (q, k layout)
// TRANS_OUT=true : out[(b*H+h)*64 + d][r]        (v stored transposed)
template <bool TRANS_OUT>
__global__ void qkv_proj_kernel(const float* __restrict__ X,
                                const float* __restrict__ W,
                                const float* __restrict__ bias,
                                _Float16* __restrict__ out, int rowsPer) {
  const int wave = threadIdx.x >> 5;
  const int n0 = (blockIdx.x * 8 + wave) * 16;
  const int m0 = blockIdx.y * 16;
  const int lane = threadIdx.x & 31;
  const int hi = lane >> 4;
  const int n = n0 + (lane & 15);

  const float* xa = X + (size_t)m0 * EMB_;
  const float* wb = W + (size_t)n0 * EMB_;
  v8f acc = {};
  for (int kk = 0; kk < EMB_; kk += 32) {
    __builtin_prefetch(xa + kk + 64, 0, 0);
    v16h a = frag_from_f32(xa + kk, EMB_);
    v16h b = frag_from_f32(wb + kk, EMB_);
    acc = wmma_step(a, b, acc);
  }
  const float bv = bias[n];
  const int h = n >> 6, d = n & 63;
#pragma unroll
  for (int r = 0; r < 8; ++r) {
    int m = m0 + hi * 8 + r;
    int bidx = m / rowsPer;
    int rin = m - bidx * rowsPer;
    size_t idx;
    if (TRANS_OUT)
      idx = ((size_t)(bidx * H_ + h) * DV_ + d) * (size_t)rowsPer + rin;
    else
      idx = ((size_t)(bidx * H_ + h) * (size_t)rowsPer + rin) * DK_ + d;
    out[idx] = (_Float16)(acc[r] + bv);
  }
}

// ---- per-row L2 norms over DK=64 ------------------------------------------
__global__ void norm_kernel(const _Float16* __restrict__ X, float* __restrict__ out,
                            int total) {
  int i = blockIdx.x * blockDim.x + threadIdx.x;
  if (i >= total) return;
  const _Float16* p = X + (size_t)i * DK_;
  float s = 0.f;
#pragma unroll
  for (int j = 0; j < DK_; ++j) { float v = (float)p[j]; s += v * v; }
  out[i] = sqrtf(s);
}

// ---- dots: att_raw = (q . k) * rcp(max(|q||k|, eps)) ----------------------
__global__ void dots_kernel(const _Float16* __restrict__ q,
                            const _Float16* __restrict__ k,
                            const float* __restrict__ qn,
                            const float* __restrict__ kn,
                            float* __restrict__ att) {
  const int wave = threadIdx.x >> 5;
  const int c0 = (blockIdx.x * 8 + wave) * 16;
  const int t0 = blockIdx.y * 16;
  const int bh = blockIdx.z;
  const int lane = threadIdx.x & 31;
  const int hi = lane >> 4;
  const int n = c0 + (lane & 15);

  const _Float16* qa = q + ((size_t)bh * NT_ + t0) * DK_;
  const _Float16* kb = k + ((size_t)bh * NC_ + c0) * DK_;
  v8f acc = {};
  acc = wmma_step(frag_from_f16(qa, DK_), frag_from_f16(kb, DK_), acc);
  acc = wmma_step(frag_from_f16(qa + 32, DK_), frag_from_f16(kb + 32, DK_), acc);

  const float knv = kn[(size_t)bh * NC_ + n];
#pragma unroll
  for (int r = 0; r < 8; ++r) {
    int t = t0 + hi * 8 + r;
    // softmax renormalizes, so v_rcp_f32 (approx, ~1 ulp) is sufficient and
    // avoids the full IEEE division expansion per element.
    float scale = __builtin_amdgcn_rcpf(
        fmaxf(qn[(size_t)bh * NT_ + t] * knv, 1e-6f));
    att[((size_t)bh * NT_ + t) * NC_ + n] = acc[r] * scale;
  }
}

// ---- softmax in place over rows of NC=2048 --------------------------------
__global__ void softmax_kernel(float* __restrict__ att) {
  float* p = att + (size_t)blockIdx.x * NC_;
  const int tid = threadIdx.x;
  float v[8];
  float mx = -3.4e38f;
#pragma unroll
  for (int i = 0; i < 8; ++i) { v[i] = p[tid + i * 256]; mx = fmaxf(mx, v[i]); }
  __shared__ float red[256];
  red[tid] = mx;
  __syncthreads();
  for (int s = 128; s > 0; s >>= 1) {
    if (tid < s) red[tid] = fmaxf(red[tid], red[tid + s]);
    __syncthreads();
  }
  mx = red[0];
  __syncthreads();
  float sum = 0.f;
#pragma unroll
  for (int i = 0; i < 8; ++i) { v[i] = __expf(v[i] - mx); sum += v[i]; }
  red[tid] = sum;
  __syncthreads();
  for (int s = 128; s > 0; s >>= 1) {
    if (tid < s) red[tid] += red[tid + s];
    __syncthreads();
  }
  const float inv = __builtin_amdgcn_rcpf(red[0]);
#pragma unroll
  for (int i = 0; i < 8; ++i) p[tid + i * 256] = v[i] * inv;
}

// ---- O = att @ V, written as (B, NT, H*DV) f16 ----------------------------
__global__ void pv_kernel(const float* __restrict__ att,     // (BH, NT, NC)
                          const _Float16* __restrict__ vT,   // (BH, 64, NC)
                          _Float16* __restrict__ O) {        // (B*NT, 512)
  const int wave = threadIdx.x >> 5;
  const int t0 = (blockIdx.y * 2 + (wave >> 2)) * 16;
  const int n0 = (wave & 3) * 16;
  const int bh = blockIdx.z;
  const int lane = threadIdx.x & 31;
  const int hi = lane >> 4;
  const int d = n0 + (lane & 15);

  const float* pa = att + ((size_t)bh * NT_ + t0) * NC_;
  const _Float16* vb = vT + ((size_t)bh * DV_ + n0) * NC_;
  v8f acc = {};
  for (int kk = 0; kk < NC_; kk += 32) {
    __builtin_prefetch(pa + kk + 64, 0, 0);
    __builtin_prefetch(vb + kk + 64, 0, 0);
    acc = wmma_step(frag_from_f32(pa + kk, NC_), frag_from_f16(vb + kk, NC_), acc);
  }
  const int b = bh >> 3, h = bh & 7;
#pragma unroll
  for (int r = 0; r < 8; ++r) {
    int t = t0 + hi * 8 + r;
    O[(size_t)(b * NT_ + t) * (H_ * DV_) + h * DV_ + d] = (_Float16)acc[r];
  }
}

// ---- final projection: out = O @ W_R^T + b_R (f32) ------------------------
__global__ void final_proj_kernel(const _Float16* __restrict__ O,  // (4096, 512)
                                  const float* __restrict__ W,     // (512, 512)
                                  const float* __restrict__ bias,
                                  float* __restrict__ out) {
  const int wave = threadIdx.x >> 5;
  const int n0 = (blockIdx.x * 8 + wave) * 16;
  const int m0 = blockIdx.y * 16;
  const int lane = threadIdx.x & 31;
  const int hi = lane >> 4;
  const int n = n0 + (lane & 15);

  v8f acc = {};
  for (int kk = 0; kk < EMB_; kk += 32) {
    v16h a = frag_from_f16(O + (size_t)m0 * EMB_ + kk, EMB_);
    v16h b = frag_from_f32(W + (size_t)n0 * EMB_ + kk, EMB_);
    acc = wmma_step(a, b, acc);
  }
  const float bv = bias[n];
#pragma unroll
  for (int r = 0; r < 8; ++r) {
    int m = m0 + hi * 8 + r;
    out[(size_t)m * EMB_ + n] = acc[r] + bv;
  }
}

extern "C" void kernel_launch(void* const* d_in, const int* in_sizes, int n_in,
                              void* d_out, int out_size, void* d_ws, size_t ws_size,
                              hipStream_t stream) {
  (void)in_sizes; (void)n_in; (void)out_size; (void)ws_size;
  const float* queries = (const float*)d_in[0];
  const float* keys    = (const float*)d_in[1];
  const float* values  = (const float*)d_in[2];
  const float* W_At = (const float*)d_in[3];
  const float* b_At = (const float*)d_in[4];
  const float* W_Ac = (const float*)d_in[5];
  const float* b_Ac = (const float*)d_in[6];
  const float* W_Bc = (const float*)d_in[7];
  const float* b_Bc = (const float*)d_in[8];
  const float* W_R  = (const float*)d_in[9];
  const float* b_R  = (const float*)d_in[10];

  float* out = (float*)d_out;                          // (B, NT, EMB)
  float* att = out + (size_t)B_ * NT_ * EMB_;          // (B, H, NT, NC)

  // workspace carve-up (~41 MB)
  _Float16* q  = (_Float16*)d_ws;                      // (BH, NT, 64)
  _Float16* k  = q  + (size_t)B_ * H_ * NT_ * DK_;     // (BH, NC, 64)
  _Float16* vT = k  + (size_t)B_ * H_ * NC_ * DK_;     // (BH, 64, NC)
  _Float16* O  = vT + (size_t)B_ * H_ * NC_ * DV_;     // (B*NT, 512)
  float* qn = (float*)(O + (size_t)B_ * NT_ * H_ * DV_);
  float* kn = qn + (size_t)B_ * H_ * NT_;

  dim3 blk(256);
  qkv_proj_kernel<false><<<dim3(4, (B_ * NT_) / 16), blk, 0, stream>>>(queries, W_At, b_At, q, NT_);
  qkv_proj_kernel<false><<<dim3(4, (B_ * NC_) / 16), blk, 0, stream>>>(keys,    W_Ac, b_Ac, k, NC_);
  qkv_proj_kernel<true ><<<dim3(4, (B_ * NC_) / 16), blk, 0, stream>>>(values,  W_Bc, b_Bc, vT, NC_);

  norm_kernel<<<(B_ * H_ * NT_) / 256, blk, 0, stream>>>(q, qn, B_ * H_ * NT_);
  norm_kernel<<<(B_ * H_ * NC_) / 256, blk, 0, stream>>>(k, kn, B_ * H_ * NC_);

  dots_kernel<<<dim3(NC_ / 128, NT_ / 16, B_ * H_), blk, 0, stream>>>(q, k, qn, kn, att);
  softmax_kernel<<<dim3(B_ * H_ * NT_), blk, 0, stream>>>(att);
  pv_kernel<<<dim3(1, NT_ / 32, B_ * H_), blk, 0, stream>>>(att, vT, O);
  final_proj_kernel<<<dim3(4, (B_ * NT_) / 16), blk, 0, stream>>>(O, W_R, b_R, out);
}
